// VectorQuantizer_13048110645555
// MI455X (gfx1250) — matrix operand order repaired and checked by reading, not compiled, not run on hardware
//
#include <hip/hip_runtime.h>
#include <hip/hip_bf16.h>
#include <stdint.h>

#define NUM_EMB 1024
#define EMB_DIM 256
#define NTOK    65536
#define TOK_WG  128
#define WSTAGE  32

typedef __attribute__((ext_vector_type(16))) __bf16          v16bf;
typedef __attribute__((ext_vector_type(8)))  float           v8f;
typedef __attribute__((ext_vector_type(8)))  unsigned short  us8;
typedef __attribute__((ext_vector_type(16))) unsigned short  us16;
typedef __attribute__((ext_vector_type(4)))  float           f4;

__device__ __forceinline__ unsigned short f2bf(float f) {
    unsigned int u = __float_as_uint(f);
    u += 0x7FFFu + ((u >> 16) & 1u);          // round-to-nearest-even
    return (unsigned short)(u >> 16);
}
__device__ __forceinline__ float bf2f(unsigned short h) {
    return __uint_as_float(((unsigned int)h) << 16);
}

// ---------------------------------------------------------------------------
// Kernel 1: codebook prep — split W into bf16 hi/lo, compute ||w_k||^2.
// grid = 1024 blocks (one per code), block = 256 threads (one per dim).
// ---------------------------------------------------------------------------
__global__ void vq_prep_w(const float* __restrict__ W,
                          unsigned short* __restrict__ Whi,
                          unsigned short* __restrict__ Wlo,
                          float* __restrict__ wn) {
    const int k = blockIdx.x;
    const int d = threadIdx.x;
    float w = W[k * EMB_DIM + d];
    unsigned short hi = f2bf(w);
    float lo = w - bf2f(hi);
    Whi[k * EMB_DIM + d] = hi;
    Wlo[k * EMB_DIM + d] = f2bf(lo);

    float s = w * w;
    #pragma unroll
    for (int m = 1; m < 32; m <<= 1) s += __shfl_xor(s, m, 32);
    __shared__ float part[8];
    const int lane = threadIdx.x & 31, wv = threadIdx.x >> 5;
    if (lane == 0) part[wv] = s;
    __syncthreads();
    if (threadIdx.x == 0) {
        float t = 0.f;
        #pragma unroll
        for (int i = 0; i < 8; ++i) t += part[i];
        wn[k] = t;
    }
}

// ---------------------------------------------------------------------------
// Kernel 2: WMMA distance argmin + codebook gather.
// 8 waves x 16-token M-tiles per workgroup. The A operand (token tile,
// bf16 hi+lo) is converted from fp32 ONCE into 128 VGPRs/lane and reused
// across all 1024 codes; only the B operand (codebook, LDS-staged 32 rows
// at a time) moves per iteration: 4 ds_load_b128 per 3 WMMAs.
// sim = Ahi*Bhi + Ahi*Blo + Alo*Bhi (fp32-accurate split product);
// dist' = ||w||^2 - 2*sim (the ||x||^2 term is argmin-invariant).
// ---------------------------------------------------------------------------
__global__ __launch_bounds__(256) void vq_argmin(
        const float* __restrict__ X,
        const unsigned short* __restrict__ Whi, const unsigned short* __restrict__ Wlo,
        const float* __restrict__ wn, const float* __restrict__ W,
        float* __restrict__ outQ, float* __restrict__ outI) {
    __shared__ unsigned short sWhi[WSTAGE * EMB_DIM];   // 16 KB
    __shared__ unsigned short sWlo[WSTAGE * EMB_DIM];   // 16 KB
    __shared__ float          sWn[NUM_EMB];             //  4 KB
    __shared__ int            sIdx[TOK_WG];

    const int tid    = threadIdx.x;
    const int wave   = tid >> 5;
    const int lane   = tid & 31;
    const int lane16 = lane & 15;
    const int half   = lane >> 4;
    const int wgBase = blockIdx.x * TOK_WG;
    const int aRow   = wave * 16 + lane16;   // token row this lane feeds

    for (int i = tid; i < NUM_EMB; i += 256) sWn[i] = wn[i];

    // ---- load this lane's half-row of X (fp32) and split to bf16 hi/lo ----
    // A layout (16x32 bf16): lane<16 holds K {kb*32+0..15};
    // lane>=16 holds K {+16..31}.
    v16bf Ahi[EMB_DIM / 32], Alo[EMB_DIM / 32];
    {
        const float* xrow = X + (size_t)(wgBase + aRow) * EMB_DIM;
        #pragma unroll
        for (int kb = 0; kb < EMB_DIM / 32; ++kb) {
            const int c0 = kb * 32 + half * 16;
            f4 fv[4];
            fv[0] = *(const f4*)(xrow + c0);
            fv[1] = *(const f4*)(xrow + c0 + 4);
            fv[2] = *(const f4*)(xrow + c0 + 8);
            fv[3] = *(const f4*)(xrow + c0 + 12);
            us16 h, l;
            #pragma unroll
            for (int q = 0; q < 4; ++q) {
                #pragma unroll
                for (int j = 0; j < 4; ++j) {
                    float x = fv[q][j];
                    unsigned short hh = f2bf(x);
                    h[q * 4 + j] = hh;
                    l[q * 4 + j] = f2bf(x - bf2f(hh));
                }
            }
            Ahi[kb] = __builtin_bit_cast(v16bf, h);
            Alo[kb] = __builtin_bit_cast(v16bf, l);
        }
    }

    float minv[8];
    int   mini[8];
    #pragma unroll
    for (int v = 0; v < 8; ++v) { minv[v] = 3.4e38f; mini[v] = 0; }

    for (int cs = 0; cs < NUM_EMB; cs += WSTAGE) {
        __syncthreads();
        {   // stage next 32 codebook rows (hi+lo) into LDS
            const us8* gh = (const us8*)(Whi + (size_t)cs * EMB_DIM);
            const us8* gl = (const us8*)(Wlo + (size_t)cs * EMB_DIM);
            us8* shh = (us8*)sWhi;
            us8* sll = (us8*)sWlo;
            #pragma unroll
            for (int r = 0; r < WSTAGE * EMB_DIM / 8 / 256; ++r) {
                const int i = r * 256 + tid;
                shh[i] = gh[i];
                sll[i] = gl[i];
            }
        }
        __syncthreads();

        #pragma unroll
        for (int c16 = 0; c16 < WSTAGE / 16; ++c16) {
            v8f acc = {0.f, 0.f, 0.f, 0.f, 0.f, 0.f, 0.f, 0.f};
            #pragma unroll
            for (int kb = 0; kb < EMB_DIM / 32; ++kb) {
                // B operand (32x16 bf16): lane group selects K-half, 16 contiguous
                const int bRow = c16 * 16 + lane16;
                const int bCol = kb * 32 + half * 16;
                us16 bh = *(const us16*)&sWhi[bRow * EMB_DIM + bCol];
                us16 bl = *(const us16*)&sWlo[bRow * EMB_DIM + bCol];
                v16bf b_hi = __builtin_bit_cast(v16bf, bh);
                v16bf b_lo = __builtin_bit_cast(v16bf, bl);

                acc = __builtin_amdgcn_wmma_f32_16x16x32_bf16(false, Ahi[kb], false, b_hi,
                                                              (short)0, acc, false, false);
                acc = __builtin_amdgcn_wmma_f32_16x16x32_bf16(false, Ahi[kb], false, b_lo,
                                                              (short)0, acc, false, false);
                acc = __builtin_amdgcn_wmma_f32_16x16x32_bf16(false, Alo[kb], false, b_hi,
                                                              (short)0, acc, false, false);
            }
            const int code = cs + c16 * 16 + lane16;   // N index is fixed per lane
            const float w2 = sWn[code];
            #pragma unroll
            for (int v = 0; v < 8; ++v) {
                float d = fmaf(-2.0f, acc[v], w2);
                if (d < minv[v]) { minv[v] = d; mini[v] = code; }
            }
        }
    }

    // reduce (min, argmin) across the 16 lanes of each half (xor stays in-half)
    #pragma unroll
    for (int v = 0; v < 8; ++v) {
        #pragma unroll
        for (int s = 1; s < 16; s <<= 1) {
            float ov = __shfl_xor(minv[v], s, 32);
            int   oi = __shfl_xor(mini[v], s, 32);
            if (ov < minv[v] || (ov == minv[v] && oi < mini[v])) { minv[v] = ov; mini[v] = oi; }
        }
    }
    // C-layout: lanes0-15 VGPR v -> M=v ; lanes16-31 -> M=v+8
    if (lane16 == 0) {
        #pragma unroll
        for (int v = 0; v < 8; ++v) sIdx[wave * 16 + half * 8 + v] = mini[v];
    }
    __syncthreads();

    // emit indices (as float, matching d_out dtype) and gather fp32 codebook rows
    for (int t = 0; t < 16; ++t) {
        const int token = wgBase + wave * 16 + t;
        const int idx   = sIdx[wave * 16 + t];
        if (lane == 0) outI[token] = (float)idx;
        const f4* src = (const f4*)(W + (size_t)idx * EMB_DIM);
        f4*       dst = (f4*)(outQ + (size_t)token * EMB_DIM);
        dst[lane]      = src[lane];        // 128 floats
        dst[lane + 32] = src[lane + 32];   // remaining 128
    }
}

// ---------------------------------------------------------------------------
extern "C" void kernel_launch(void* const* d_in, const int* in_sizes, int n_in,
                              void* d_out, int out_size, void* d_ws, size_t ws_size,
                              hipStream_t stream) {
    const float* X = (const float*)d_in[0];   // [64,32,32,256] fp32
    const float* W = (const float*)d_in[1];   // [1024,256] fp32
    float* outQ = (float*)d_out;                         // quantized (N*D floats)
    float* outI = outQ + (size_t)NTOK * EMB_DIM;         // indices as floats (N)

    char* ws = (char*)d_ws;
    unsigned short* Whi = (unsigned short*)ws; ws += (size_t)NUM_EMB * EMB_DIM * 2;
    unsigned short* Wlo = (unsigned short*)ws; ws += (size_t)NUM_EMB * EMB_DIM * 2;
    float*          wnm = (float*)ws;

    vq_prep_w<<<NUM_EMB, EMB_DIM, 0, stream>>>(W, Whi, Wlo, wnm);
    vq_argmin<<<NTOK / TOK_WG, 256, 0, stream>>>(X, Whi, Wlo, wnm, W, outQ, outI);
}
